// BlocksCore_67035849556272
// MI455X (gfx1250) — matrix operand order repaired and checked by reading, not compiled, not run on hardware
//
#include <hip/hip_runtime.h>
#include <hip/hip_bf16.h>
#include <math.h>

// ---------------------------------------------------------------------------
// Types for CDNA5 WMMA
// ---------------------------------------------------------------------------
typedef _Float16 v16h __attribute__((ext_vector_type(16)));
typedef _Float16 v8h  __attribute__((ext_vector_type(8)));
typedef _Float16 v2h  __attribute__((ext_vector_type(2)));
typedef float    v8f  __attribute__((ext_vector_type(8)));

#define B_SZ     1024
#define NB_IN    4
#define NB_OUT   8
#define BS_IN    128
#define NHID     4096
#define BS_OUT   512      // NHID / NB_OUT
#define DK       64
#define NGATE    2048     // 4 * BS_OUT

__device__ __forceinline__ float sigf(float x) { return 1.0f / (1.0f + expf(-x)); }

// ---------------------------------------------------------------------------
// Generic grouped GEMM:  C[g](M x N) (+)= A[:, g*a_goff + k] @ W[g](K x N) (+bias)
//   Template NACC: accumulators per wave; workgroup tile = 64 x (NACC*32).
//   grid = (M/64, N/(NACC*32), groups), block = 256 (8 waves).
//   Wave w: rows (w&3)*16..+15, cols (w>>2)*(NACC*16) .. +NACC*16-1.
//   K multiple of 32; M multiple of 64; N multiple of NACC*32.
//   f32 inputs are converted to packed f16 while staged into LDS (v2h b32
//   stores); fragments read with aligned ds_load_b128; accumulate in f32 via
//   v_wmma_f32_16x16x32_f16.
// ---------------------------------------------------------------------------
template <int NACC>
__global__ __launch_bounds__(256)
void gemm_wmma_f16(const float* __restrict__ A, int lda, int a_goff,
                   const float* __restrict__ W, int w_gstride,
                   float* __restrict__ C, int ldc, int c_goff,
                   const float* __restrict__ bias,
                   int K, int N, int accumulate)
{
    constexpr int NT = NACC * 32;                      // workgroup tile width
    const int g     = blockIdx.z;
    const int row0  = blockIdx.x * 64;
    const int ncol0 = blockIdx.y * NT;                 // within-group col base
    const float* Wg = W + (size_t)g * w_gstride;
    const int acol0 = g * a_goff;
    const int ccol0 = g * c_goff + ncol0;

    // LDS tiles, f16, row stride 40 halves (80 B -> 16B-aligned b128 loads)
    __shared__ __align__(16) _Float16 ldsA[64 * 40];   // [row][k]   row<64,k<32
    __shared__ __align__(16) _Float16 ldsB[NT * 40];   // [col][k]   col<NT,k<32

    const int tid  = threadIdx.x;
    const int lane = tid & 31;
    const int wave = tid >> 5;
    const int wm   = wave & 3;     // 0..3 row subtile
    const int wn   = wave >> 2;    // 0..1 col subtile (NACC*16 wide)

    v8f acc[NACC];
    #pragma unroll
    for (int s = 0; s < NACC; ++s) acc[s] = v8f{};

    for (int k0 = 0; k0 < K; k0 += 32) {
        // Stage A tile: 64 rows x 32 K; f32x2 -> packed f16, b32 LDS stores.
        // 1024 pairs / 256 threads = 4 pairs each; coalesced b64 global loads.
        #pragma unroll
        for (int i = 0; i < 4; ++i) {
            int e   = tid + i * 256;       // pair index
            int r   = e >> 4;
            int kkp = (e & 15) * 2;
            const float2 f =
                *(const float2*)&A[(size_t)(row0 + r) * lda + acol0 + k0 + kkp];
            v2h p = { (_Float16)f.x, (_Float16)f.y };
            *(v2h*)&ldsA[r * 40 + kkp] = p;
        }
        // Stage B tile: 32 K x NT cols stored transposed [col][k]; pack two
        // K-adjacent rows per b32 store; loads coalesced over columns.
        #pragma unroll
        for (int i = 0; i < NT / 16; ++i) {
            int e  = tid + i * 256;        // pair index over (16 kp x NT cols)
            int c  = e % NT;
            int kk = (e / NT) * 2;
            float f0 = Wg[(size_t)(k0 + kk) * N + ncol0 + c];
            float f1 = Wg[(size_t)(k0 + kk + 1) * N + ncol0 + c];
            v2h p = { (_Float16)f0, (_Float16)f1 };
            *(v2h*)&ldsB[c * 40 + kk] = p;
        }
        __syncthreads();

        const int half = lane >> 4;        // 0: lanes 0-15, 1: lanes 16-31
        const int mloc = lane & 15;
        const int nloc = lane & 15;

        // A fragment (ISA 16-bit A 16x32 layout):
        //   half=0: K = h (h<8), 16+(h-8) (h>=8); half=1: +8 on each run.
        const _Float16* ap = &ldsA[(wm * 16 + mloc) * 40];
        v8h alo = *(const v8h*)(ap + 8 * half);        // K = 8*half + 0..7
        v8h ahi = *(const v8h*)(ap + 16 + 8 * half);   // K = 16 + 8*half + 0..7
        v16h afrag;
        #pragma unroll
        for (int i = 0; i < 8; ++i) { afrag[i] = alo[i]; afrag[i + 8] = ahi[i]; }

        // B fragments (same K order as A: K = 8*half + 0..7, 16 + 8*half + 0..7)
        #pragma unroll
        for (int s = 0; s < NACC; ++s) {
            const _Float16* bp =
                &ldsB[(wn * (NACC * 16) + s * 16 + nloc) * 40 + 8 * half];
            v8h blo = *(const v8h*)(bp);
            v8h bhi = *(const v8h*)(bp + 16);
            v16h bfrag;
            #pragma unroll
            for (int i = 0; i < 8; ++i) { bfrag[i] = blo[i]; bfrag[i + 8] = bhi[i]; }
            acc[s] = __builtin_amdgcn_wmma_f32_16x16x32_f16(
                false, afrag, false, bfrag, (short)0, acc[s], false, false);
        }
        __syncthreads();
    }

    // Store C (f32 16x16 C/D layout: VGPR r -> M = r + 8*(lane/16), N = lane%16)
    const int half = lane >> 4;
    const int nl   = lane & 15;
    #pragma unroll
    for (int s = 0; s < NACC; ++s) {
        #pragma unroll
        for (int r = 0; r < 8; ++r) {
            int row = row0 + wm * 16 + r + 8 * half;
            int c   = wn * (NACC * 16) + s * 16 + nl;
            size_t idx = (size_t)row * ldc + ccol0 + c;
            float v = acc[s][r];
            if (bias) v += bias[ncol0 + c];
            if (accumulate) v += C[idx];
            C[idx] = v;
        }
    }
}

// ---------------------------------------------------------------------------
// Attention over key blocks. One wave per (b, n) pair; lane owns 2 of 64 dims.
// EXTRA_ZERO adds an implicit all-zero key/value block (logit 0, value 0) —
// the padded 5th input block of the reference.
// ---------------------------------------------------------------------------
template <int NK, bool EXTRA_ZERO>
__global__ __launch_bounds__(256)
void attn_kernel(const float* __restrict__ q, const float* __restrict__ k,
                 const float* __restrict__ v, float* __restrict__ o,
                 float* __restrict__ score, int qld, int kld)
{
    const int wid  = blockIdx.x * 8 + (threadIdx.x >> 5);   // 0..B*NB_OUT-1
    const int lane = threadIdx.x & 31;
    const int b    = wid >> 3;
    const int n    = wid & 7;
    const int d0   = lane * 2;

    const float* qp = q + (size_t)b * qld + n * DK;
    const float  q0 = qp[d0];
    const float  q1 = qp[d0 + 1];
    const float* kb = k + (size_t)b * kld;
    const float* vb = v + (size_t)b * kld;

    float logits[NK + 1];
    #pragma unroll
    for (int j = 0; j < NK; ++j) {
        float p = q0 * kb[j * DK + d0] + q1 * kb[j * DK + d0 + 1];
        #pragma unroll
        for (int off = 16; off; off >>= 1) p += __shfl_xor(p, off, 32);
        logits[j] = p * 0.125f;                     // 1/sqrt(DK)
    }
    const int cnt = NK + (EXTRA_ZERO ? 1 : 0);
    if (EXTRA_ZERO) logits[NK] = 0.0f;

    float mx = logits[0];
    #pragma unroll
    for (int j = 1; j < NK + 1; ++j) if (j < cnt) mx = fmaxf(mx, logits[j]);
    float wsum = 0.0f;
    float wj[NK + 1];
    #pragma unroll
    for (int j = 0; j < NK + 1; ++j) {
        if (j < cnt) { wj[j] = expf(logits[j] - mx); wsum += wj[j]; }
    }
    const float inv = 1.0f / wsum;

    float o0 = 0.0f, o1 = 0.0f;
    #pragma unroll
    for (int j = 0; j < NK; ++j) {
        float a = wj[j] * inv;
        o0 += a * vb[j * DK + d0];
        o1 += a * vb[j * DK + d0 + 1];
    }
    float* op = o + (size_t)wid * DK;
    op[d0]     = o0;
    op[d0 + 1] = o1;
    if (EXTRA_ZERO && score && lane == 0) score[wid] = wj[0] * inv;
}

// ---------------------------------------------------------------------------
// Top-4-of-8 block mask (one thread per batch row)
// ---------------------------------------------------------------------------
__global__ void topk_mask_kernel(const float* __restrict__ score,
                                 float* __restrict__ maskblk)
{
    int b = blockIdx.x * blockDim.x + threadIdx.x;
    if (b >= B_SZ) return;
    float s[NB_OUT], a[NB_OUT];
    #pragma unroll
    for (int i = 0; i < NB_OUT; ++i) { s[i] = score[b * NB_OUT + i]; a[i] = s[i]; }
    #pragma unroll
    for (int it = 0; it < 4; ++it) {          // partial selection sort
        int mi = it;
        #pragma unroll
        for (int j = 0; j < NB_OUT; ++j)
            if (j > it && a[j] > a[mi]) mi = j;
        float t = a[it]; a[it] = a[mi]; a[mi] = t;
    }
    const float thr = a[3] - 0.01f;
    #pragma unroll
    for (int i = 0; i < NB_OUT; ++i)
        maskblk[b * NB_OUT + i] = (s[i] > thr) ? 1.0f : 0.0f;
}

// ---------------------------------------------------------------------------
// LSTM pointwise: gates (B x NB_OUT x 2048) -> h_new, c_new (B x NHID)
// ---------------------------------------------------------------------------
__global__ __launch_bounds__(256)
void lstm_kernel(const float* __restrict__ gates, const float* __restrict__ cx,
                 const float* __restrict__ b_ih, const float* __restrict__ b_hh,
                 float* __restrict__ h_new, float* __restrict__ c_new)
{
    const int idx = blockIdx.x * 256 + threadIdx.x;    // < B*NHID
    const int b   = idx >> 12;
    const int rem = idx & (NHID - 1);
    const int k   = rem >> 9;
    const int d   = rem & (BS_OUT - 1);
    const size_t gbase = (size_t)b * (NB_OUT * NGATE) + k * NGATE;
    const int    bb    = k * NGATE;

    float ig = gates[gbase + d]        + b_ih[bb + d]        + b_hh[bb + d];
    float fg = gates[gbase + 512 + d]  + b_ih[bb + 512 + d]  + b_hh[bb + 512 + d];
    float gg = gates[gbase + 1024 + d] + b_ih[bb + 1024 + d] + b_hh[bb + 1024 + d];
    float og = gates[gbase + 1536 + d] + b_ih[bb + 1536 + d] + b_hh[bb + 1536 + d];

    float c = sigf(fg) * cx[idx] + sigf(ig) * tanhf(gg);
    float h = sigf(og) * tanhf(c);
    c_new[idx] = c;
    h_new[idx] = h;
}

// ---------------------------------------------------------------------------
// Final gated residual + block mask; writes hx_out | cx_out | mask to d_out
// ---------------------------------------------------------------------------
__global__ __launch_bounds__(256)
void final_kernel(const float* __restrict__ outb, const float* __restrict__ gateb,
                  const float* __restrict__ h_new, const float* __restrict__ c_new,
                  const float* __restrict__ maskblk,
                  const float* __restrict__ hx, const float* __restrict__ cx,
                  float* __restrict__ out)
{
    const int idx = blockIdx.x * 256 + threadIdx.x;    // < B*NHID
    const int b   = idx >> 12;
    const int rem = idx & (NHID - 1);
    const int k   = rem >> 9;

    const float g    = sigf(gateb[idx]);               // bias added in GEMM
    const float hxn  = h_new[idx] + g * tanhf(outb[idx]);
    const float m    = maskblk[b * NB_OUT + k];
    out[idx]                   = m * hxn        + (1.0f - m) * hx[idx];
    out[B_SZ * NHID + idx]     = m * c_new[idx] + (1.0f - m) * cx[idx];
    out[2 * B_SZ * NHID + idx] = m;
}

// ---------------------------------------------------------------------------
// Host-side orchestration
// ---------------------------------------------------------------------------
extern "C" void kernel_launch(void* const* d_in, const int* in_sizes, int n_in,
                              void* d_out, int out_size, void* d_ws, size_t ws_size,
                              hipStream_t stream)
{
    const float* inp        = (const float*)d_in[0];
    const float* hx         = (const float*)d_in[1];
    const float* cx         = (const float*)d_in[2];
    const float* ia_wq      = (const float*)d_in[3];
    const float* ia_wk      = (const float*)d_in[4];
    const float* ia_wv      = (const float*)d_in[5];
    const float* ia_fc_w    = (const float*)d_in[6];
    const float* ia_fc_b    = (const float*)d_in[7];
    const float* mha_wq     = (const float*)d_in[8];
    const float* mha_wk     = (const float*)d_in[9];
    const float* mha_wv     = (const float*)d_in[10];
    const float* mha_fc_w   = (const float*)d_in[11];
    const float* mha_fc_b   = (const float*)d_in[12];
    const float* mha_gate_w = (const float*)d_in[13];
    const float* mha_gate_b = (const float*)d_in[14];
    const float* w_ih       = (const float*)d_in[15];
    const float* w_hh       = (const float*)d_in[16];
    const float* b_ih       = (const float*)d_in[17];
    const float* b_hh       = (const float*)d_in[18];

    float* ws = (float*)d_ws;
    // Workspace layout (floats); later stages alias buffers dead by then.
    float* inp_use = ws;                              // B*NHID          = 4194304
    float* gates   = inp_use + 4194304;               // B*NB_OUT*2048   = 16777216
    float* h_new   = gates + 16777216;                // 4194304
    float* c_new   = h_new + 4194304;                 // 4194304
    float* maskblk = c_new + 4194304;                 // 8192
    float* scorebf = maskblk + 8192;                  // 8192
    float* qbuf    = scorebf + 8192;                  // 524288 (aliased by q2)
    float* kbuf    = qbuf + 524288;                   // 262144
    float* vbuf    = kbuf + 262144;                   // 262144
    float* obuf    = vbuf + 262144;                   // 524288 (aliased by o2)
    float* k2buf   = obuf + 524288;                   // 524288
    float* v2buf   = k2buf + 524288;                  // 524288
    float* q2buf   = qbuf;     // q dead after attn1
    float* o2buf   = obuf;     // o dead after inp_use GEMM
    float* outb    = gates;    // gates dead after lstm_kernel
    float* gateb   = gates + 4194304;

    dim3 blk(256);

    // --- input attention projections ---------------------------------------
    // q = hb @ ia_wq  : (1024x512)@(512x64) x 8 groups
    gemm_wmma_f16<2><<<dim3(16, 1, 8), blk, 0, stream>>>(
        hx, NHID, BS_OUT, ia_wq, BS_OUT * DK, qbuf, NB_OUT * DK, DK,
        nullptr, BS_OUT, DK, 0);
    // k,v = x @ ia_wk/wv : (1024x128)@(128x64) x 4 real groups (block 4 is 0)
    gemm_wmma_f16<2><<<dim3(16, 1, 4), blk, 0, stream>>>(
        inp, NB_IN * BS_IN, BS_IN, ia_wk, BS_IN * DK, kbuf, NB_IN * DK, DK,
        nullptr, BS_IN, DK, 0);
    gemm_wmma_f16<2><<<dim3(16, 1, 4), blk, 0, stream>>>(
        inp, NB_IN * BS_IN, BS_IN, ia_wv, BS_IN * DK, vbuf, NB_IN * DK, DK,
        nullptr, BS_IN, DK, 0);

    // --- attention 1 (4 real k-blocks + implicit zero block) + score -------
    attn_kernel<4, true><<<dim3(B_SZ * NB_OUT / 8), blk, 0, stream>>>(
        qbuf, kbuf, vbuf, obuf, scorebf, NB_OUT * DK, NB_IN * DK);

    // --- top-4 mask ---------------------------------------------------------
    topk_mask_kernel<<<dim3(4), blk, 0, stream>>>(scorebf, maskblk);

    // --- inp_use = o @ ia_fc_w + b : (8192x64)@(64x512) ---------------------
    gemm_wmma_f16<4><<<dim3(128, 4, 1), blk, 0, stream>>>(
        obuf, DK, 0, ia_fc_w, 0, inp_use, BS_OUT, 0,
        ia_fc_b, DK, BS_OUT, 0);

    // --- LSTM gate GEMMs: xb@w_ih then += hb@w_hh ---------------------------
    gemm_wmma_f16<4><<<dim3(16, 16, 8), blk, 0, stream>>>(
        inp_use, NHID, BS_OUT, w_ih, BS_OUT * NGATE, gates, NB_OUT * NGATE, NGATE,
        nullptr, BS_OUT, NGATE, 0);
    gemm_wmma_f16<4><<<dim3(16, 16, 8), blk, 0, stream>>>(
        hx, NHID, BS_OUT, w_hh, BS_OUT * NGATE, gates, NB_OUT * NGATE, NGATE,
        nullptr, BS_OUT, NGATE, 1);

    // --- LSTM pointwise -----------------------------------------------------
    lstm_kernel<<<dim3(B_SZ * NHID / 256), blk, 0, stream>>>(
        gates, cx, b_ih, b_hh, h_new, c_new);

    // --- output attention projections: (1024x512)@(512x64) x 8 --------------
    gemm_wmma_f16<2><<<dim3(16, 1, 8), blk, 0, stream>>>(
        h_new, NHID, BS_OUT, mha_wq, BS_OUT * DK, q2buf, NB_OUT * DK, DK,
        nullptr, BS_OUT, DK, 0);
    gemm_wmma_f16<2><<<dim3(16, 1, 8), blk, 0, stream>>>(
        h_new, NHID, BS_OUT, mha_wk, BS_OUT * DK, k2buf, NB_OUT * DK, DK,
        nullptr, BS_OUT, DK, 0);
    gemm_wmma_f16<2><<<dim3(16, 1, 8), blk, 0, stream>>>(
        h_new, NHID, BS_OUT, mha_wv, BS_OUT * DK, v2buf, NB_OUT * DK, DK,
        nullptr, BS_OUT, DK, 0);

    // --- attention 2 (8 k-blocks) -------------------------------------------
    attn_kernel<8, false><<<dim3(B_SZ * NB_OUT / 8), blk, 0, stream>>>(
        q2buf, k2buf, v2buf, o2buf, nullptr, NB_OUT * DK, NB_OUT * DK);

    // --- out / gate projections: (8192x64)@(64x512) --------------------------
    gemm_wmma_f16<4><<<dim3(128, 4, 1), blk, 0, stream>>>(
        o2buf, DK, 0, mha_fc_w, 0, outb, BS_OUT, 0,
        mha_fc_b, DK, BS_OUT, 0);
    gemm_wmma_f16<4><<<dim3(128, 4, 1), blk, 0, stream>>>(
        o2buf, DK, 0, mha_gate_w, 0, gateb, BS_OUT, 0,
        mha_gate_b, DK, BS_OUT, 0);

    // --- final gated residual + mask ----------------------------------------
    final_kernel<<<dim3(B_SZ * NHID / 256), blk, 0, stream>>>(
        outb, gateb, h_new, c_new, maskblk, hx, cx, (float*)d_out);
}
